// SamVisionAttentionSplit_45329084842402
// MI455X (gfx1250) — matrix operand order repaired and checked by reading, not compiled, not run on hardware
//
#include <hip/hip_runtime.h>

typedef unsigned short u16;
typedef __attribute__((ext_vector_type(16))) __bf16 v16bf;
typedef __attribute__((ext_vector_type(8)))  __bf16 v8bf;
typedef __attribute__((ext_vector_type(8)))  float  v8f;
typedef __attribute__((ext_vector_type(4)))  int    v4i;

union V16 { v16bf v; v8bf h[2]; };

__device__ __forceinline__ float bf2f(u16 u) {
  unsigned int x = ((unsigned int)u) << 16;
  return __builtin_bit_cast(float, x);
}
__device__ __forceinline__ u16 f2bf(float f) {
  unsigned int x = __builtin_bit_cast(unsigned int, f);
  x += 0x7FFFu + ((x >> 16) & 1u);
  return (u16)(x >> 16);
}
__device__ __forceinline__ v8f wmma_bf16(v16bf a, v16bf b, v8f c) {
  return __builtin_amdgcn_wmma_f32_16x16x32_bf16(false, a, false, b, (short)0, c, false, false);
}

// ---- CDNA5 async memory->LDS (ASYNCcnt) -------------------------------------
#define AS_GLOBAL __attribute__((address_space(1)))
#define AS_LDS    __attribute__((address_space(3)))

__device__ __forceinline__ void async_b128(const u16* g, u16* l) {
#if __has_builtin(__builtin_amdgcn_global_load_async_to_lds_b128)
  __builtin_amdgcn_global_load_async_to_lds_b128(
      (AS_GLOBAL v4i*)g, (AS_LDS v4i*)l, 0, 0);
#else
  // synchronous fallback: 16B through VGPRs
  *reinterpret_cast<uint4*>(l) = *reinterpret_cast<const uint4*>(g);
#endif
}

template <int N>
__device__ __forceinline__ void wait_async() {
#if __has_builtin(__builtin_amdgcn_global_load_async_to_lds_b128)
  asm volatile("s_wait_asynccnt %0" ::"n"(N) : "memory");
#endif
}

// ---------------------------------------------------------------- convert
__global__ void k_cvt(const float* __restrict__ src, u16* __restrict__ dst, int n) {
  int i = blockIdx.x * blockDim.x + threadIdx.x;
  if (i < n) dst[i] = f2bf(src[i]);
}

// ---------------------------------------------------------------- QKV GEMM
// out[m,n] = sum_k x[m,k] * W[n,k] + bias[n]
// z=0: Q -> qb[m*768+n]; z=1: K -> kb[m*768+n]; z=2: V -> vt[n*4096+m] (transposed)
__global__ __launch_bounds__(128) void k_qkv(
    const u16* __restrict__ xb,
    const u16* __restrict__ wq, const u16* __restrict__ wk, const u16* __restrict__ wv,
    const float* __restrict__ qbias, const float* __restrict__ kbias, const float* __restrict__ vbias,
    u16* __restrict__ qb, u16* __restrict__ kb, u16* __restrict__ vt) {
  const int mt = blockIdx.x, nt = blockIdx.y, z = blockIdx.z;
  const u16* W = (z == 0) ? wq : (z == 1) ? wk : wv;
  const float* bias = (z == 0) ? qbias : (z == 1) ? kbias : vbias;
  const int lane = threadIdx.x & 31, wave = threadIdx.x >> 5;
  const int half = lane >> 4, l16 = lane & 15;

  const int mA = mt * 64 + wave * 16 + l16;        // A-matrix row for this lane
  v8f acc[4] = {};
#pragma unroll 4
  for (int ks = 0; ks < 24; ++ks) {
    const int c0 = ks * 32;
    V16 a;
    a.h[0] = *reinterpret_cast<const v8bf*>(xb + (size_t)mA * 768 + c0 + 16 * half);
    a.h[1] = *reinterpret_cast<const v8bf*>(xb + (size_t)mA * 768 + c0 + 16 * half + 8);
#pragma unroll
    for (int j = 0; j < 4; ++j) {
      const int n = nt * 64 + j * 16 + l16;
      const v16bf b = *reinterpret_cast<const v16bf*>(W + (size_t)n * 768 + c0 + 16 * half);
      acc[j] = wmma_bf16(a.v, b, acc[j]);
    }
  }
#pragma unroll
  for (int j = 0; j < 4; ++j) {
    const int n = nt * 64 + j * 16 + l16;
    const float bv = bias[n];
#pragma unroll
    for (int g = 0; g < 8; ++g) {
      const int mm = mt * 64 + wave * 16 + 8 * half + g;   // C layout: M = g + 8*half
      const u16 hv = f2bf(acc[j][g] + bv);
      if (z == 2)       vt[(size_t)n * 4096 + mm] = hv;
      else if (z == 0)  qb[(size_t)mm * 768 + n] = hv;
      else              kb[(size_t)mm * 768 + n] = hv;
    }
  }
}

// ---------------------------------------------------------------- rel-pos bias
__global__ __launch_bounds__(64) void k_rel(
    const u16* __restrict__ qb, const float* __restrict__ rph, const float* __restrict__ rpw,
    float* __restrict__ relh, float* __restrict__ relw) {
  const int qh = blockIdx.x, h = blockIdx.y;
  const int qw = threadIdx.x;
  float q[64];
  const u16* qr = qb + ((size_t)(qh * 64 + qw)) * 768 + h * 64;
#pragma unroll
  for (int c = 0; c < 64; ++c) q[c] = bf2f(qr[c]);
  const size_t base = ((size_t)(h * 64 + qh)) * 4096 + (size_t)qw * 64;
  for (int t = 0; t < 64; ++t) {
    const float* r = rph + (size_t)(qh - t + 63) * 64;
    float s = 0.f;
#pragma unroll
    for (int c = 0; c < 64; ++c) s += q[c] * r[c];
    relh[base + t] = s;
  }
  for (int kw = 0; kw < 64; ++kw) {
    const float* r = rpw + (size_t)(qw - kw + 63) * 64;
    float s = 0.f;
#pragma unroll
    for (int c = 0; c < 64; ++c) s += q[c] * r[c];
    relw[base + kw] = s;
  }
}

// ---------------------------------------------------------------- flash attention
// One block per (qh, head): 64 queries, 64 key tiles. K/V tiles double-buffered in
// LDS via global_load_async_to_lds_b128 (ASYNCcnt), shared by all 4 waves.
__global__ __launch_bounds__(128) void k_attn(
    const u16* __restrict__ qb, const u16* __restrict__ kb, const u16* __restrict__ vt,
    const float* __restrict__ relh, const float* __restrict__ relw,
    u16* __restrict__ ao) {
  __shared__ float s_relh[64 * 64];
  __shared__ float s_relw[64 * 64];
  __shared__ u16   s_p[4][16 * 64];
  __shared__ u16   s_k[2][64 * 64];   // [key_local][c]
  __shared__ u16   s_v[2][64 * 64];   // [d][key_local]

  const int qh = blockIdx.x, h = blockIdx.y;
  const int tid = threadIdx.x;
  const int lane = tid & 31, wave = tid >> 5;
  const int half = lane >> 4, l16 = lane & 15;

  const size_t relbase = ((size_t)(h * 64 + qh)) * 4096;
  for (int i = tid; i < 4096; i += 128) {
    s_relh[i] = relh[relbase + i];
    s_relw[i] = relw[relbase + i];
  }

  // cooperative async fill of one K/V tile pair (8 async ops per wave)
  auto issue_tile = [&](int t, int buf) {
#pragma unroll
    for (int r = 0; r < 4; ++r) {
      const int ch = tid + r * 128;            // 0..511 chunk id (16B chunks)
      const int row = ch >> 3, c8 = (ch & 7) * 8;
      async_b128(kb + (size_t)(t * 64 + row) * 768 + h * 64 + c8,
                 &s_k[buf][row * 64 + c8]);
      async_b128(vt + (size_t)(h * 64 + row) * 4096 + t * 64 + c8,
                 &s_v[buf][row * 64 + c8]);
    }
  };

  // Q A-frags (persist for the whole kernel), rows = wave's 16-query strip
  const int mq = qh * 64 + wave * 16 + l16;
  const u16* qrow = qb + (size_t)mq * 768 + h * 64;
  V16 qa[2];
#pragma unroll
  for (int ks = 0; ks < 2; ++ks) {
    qa[ks].h[0] = *reinterpret_cast<const v8bf*>(qrow + ks * 32 + 16 * half);
    qa[ks].h[1] = *reinterpret_cast<const v8bf*>(qrow + ks * 32 + 16 * half + 8);
  }

  float mrun[8], lrun[8];
#pragma unroll
  for (int g = 0; g < 8; ++g) { mrun[g] = -1e30f; lrun[g] = 0.f; }
  v8f o[4] = {};
  const float scale = 0.125f;  // hd=64 -> 1/sqrt(64)

  issue_tile(0, 0);

  for (int t = 0; t < 64; ++t) {
    const int cur = t & 1;
    if (t + 1 < 64) {
      issue_tile(t + 1, cur ^ 1);
      wait_async<8>();   // current tile's 8 ops complete (in-order), next 8 in flight
    } else {
      wait_async<0>();
    }
    __syncthreads();     // tiles visible to all waves (also covers rel-bias staging @t=0)

    // ---- S = Q K^T for this key tile (B-frags from LDS)
    v8f s[4];
#pragma unroll
    for (int j = 0; j < 4; ++j) {
      v8f c = {};
#pragma unroll
      for (int ks = 0; ks < 2; ++ks) {
        const v16bf b = *reinterpret_cast<const v16bf*>(
            &s_k[cur][(j * 16 + l16) * 64 + ks * 32 + 16 * half]);
        c = wmma_bf16(qa[ks].v, b, c);
      }
      s[j] = c;
    }
    // ---- scale + rel-pos bias, row max
    float mx[8];
#pragma unroll
    for (int g = 0; g < 8; ++g) {
      const int row = wave * 16 + 8 * half + g;   // local query index (0..63)
      const float rh = s_relh[row * 64 + t];
      float m = -1e30f;
#pragma unroll
      for (int j = 0; j < 4; ++j) {
        const int col = j * 16 + l16;
        float v = s[j][g] * scale + rh + s_relw[row * 64 + col];
        s[j][g] = v;
        m = fmaxf(m, v);
      }
      mx[g] = m;
    }
#pragma unroll
    for (int g = 0; g < 8; ++g)
#pragma unroll
      for (int off = 8; off >= 1; off >>= 1)
        mx[g] = fmaxf(mx[g], __shfl_xor(mx[g], off, 32));
    // ---- online softmax update
    float corr[8], rs[8];
#pragma unroll
    for (int g = 0; g < 8; ++g) {
      const float mnew = fmaxf(mrun[g], mx[g]);
      corr[g] = __expf(mrun[g] - mnew);
      mrun[g] = mnew;
      float sum = 0.f;
#pragma unroll
      for (int j = 0; j < 4; ++j) {
        const float p = __expf(s[j][g] - mnew);
        s[j][g] = p;
        sum += p;
      }
      rs[g] = sum;
    }
#pragma unroll
    for (int g = 0; g < 8; ++g)
#pragma unroll
      for (int off = 8; off >= 1; off >>= 1)
        rs[g] += __shfl_xor(rs[g], off, 32);
#pragma unroll
    for (int g = 0; g < 8; ++g) lrun[g] = lrun[g] * corr[g] + rs[g];
#pragma unroll
    for (int j = 0; j < 4; ++j)
#pragma unroll
      for (int g = 0; g < 8; ++g) o[j][g] *= corr[g];

    // ---- P: C layout -> per-wave LDS -> A layout (same-wave DS ops are in order)
#pragma unroll
    for (int j = 0; j < 4; ++j)
#pragma unroll
      for (int g = 0; g < 8; ++g) {
        const int row = 8 * half + g;
        const int col = j * 16 + l16;
        s_p[wave][row * 64 + col] = f2bf(s[j][g]);
      }
    const u16* pp = &s_p[wave][0];
#pragma unroll
    for (int ks = 0; ks < 2; ++ks) {
      V16 pa;
      pa.h[0] = *reinterpret_cast<const v8bf*>(pp + l16 * 64 + ks * 32 + 16 * half);
      pa.h[1] = *reinterpret_cast<const v8bf*>(pp + l16 * 64 + ks * 32 + 16 * half + 8);
#pragma unroll
      for (int j = 0; j < 4; ++j) {
        const v16bf b = *reinterpret_cast<const v16bf*>(
            &s_v[cur][(j * 16 + l16) * 64 + ks * 32 + 16 * half]);
        o[j] = wmma_bf16(pa.v, b, o[j]);
      }
    }
    __syncthreads();     // all waves done reading buf 'cur' before it is refilled
  }

  // ---- normalize and emit per-head output (bf16, channel-interleaved layout)
  float inv[8];
#pragma unroll
  for (int g = 0; g < 8; ++g) inv[g] = 1.f / lrun[g];
#pragma unroll
  for (int j = 0; j < 4; ++j)
#pragma unroll
    for (int g = 0; g < 8; ++g) {
      const int mm = qh * 64 + wave * 16 + 8 * half + g;
      const int n = h * 64 + j * 16 + l16;
      ao[(size_t)mm * 768 + n] = f2bf(o[j][g] * inv[g]);
    }
}

// ---------------------------------------------------------------- output projection
__global__ __launch_bounds__(128) void k_proj(
    const u16* __restrict__ ab, const u16* __restrict__ wp,
    const float* __restrict__ bias, float* __restrict__ out) {
  const int mt = blockIdx.x, nt = blockIdx.y;
  const int lane = threadIdx.x & 31, wave = threadIdx.x >> 5;
  const int half = lane >> 4, l16 = lane & 15;
  const int mA = mt * 64 + wave * 16 + l16;
  v8f acc[4] = {};
#pragma unroll 4
  for (int ks = 0; ks < 24; ++ks) {
    const int c0 = ks * 32;
    V16 a;
    a.h[0] = *reinterpret_cast<const v8bf*>(ab + (size_t)mA * 768 + c0 + 16 * half);
    a.h[1] = *reinterpret_cast<const v8bf*>(ab + (size_t)mA * 768 + c0 + 16 * half + 8);
#pragma unroll
    for (int j = 0; j < 4; ++j) {
      const int n = nt * 64 + j * 16 + l16;
      const v16bf b = *reinterpret_cast<const v16bf*>(wp + (size_t)n * 768 + c0 + 16 * half);
      acc[j] = wmma_bf16(a.v, b, acc[j]);
    }
  }
#pragma unroll
  for (int j = 0; j < 4; ++j) {
    const int n = nt * 64 + j * 16 + l16;
    const float bv = bias[n];
#pragma unroll
    for (int g = 0; g < 8; ++g) {
      const int mm = mt * 64 + wave * 16 + 8 * half + g;
      out[(size_t)mm * 768 + n] = acc[j][g] + bv;
    }
  }
}

// ---------------------------------------------------------------- host
extern "C" void kernel_launch(void* const* d_in, const int* in_sizes, int n_in,
                              void* d_out, int out_size, void* d_ws, size_t ws_size,
                              hipStream_t stream) {
  (void)in_sizes; (void)n_in; (void)out_size; (void)ws_size;
  const float* x   = (const float*)d_in[0];
  const float* q_w = (const float*)d_in[1];
  const float* q_b = (const float*)d_in[2];
  const float* k_w = (const float*)d_in[3];
  const float* k_b = (const float*)d_in[4];
  const float* v_w = (const float*)d_in[5];
  const float* v_b = (const float*)d_in[6];
  const float* p_w = (const float*)d_in[7];
  const float* p_b = (const float*)d_in[8];
  const float* rph = (const float*)d_in[9];
  const float* rpw = (const float*)d_in[10];

  char* w = (char*)d_ws;
  size_t off = 0;
  auto take = [&](size_t bytes) -> void* {
    void* p = w + off;
    off += (bytes + 255) & ~(size_t)255;
    return p;
  };
  const size_t NTOK = 4096, C = 768;
  u16* xb  = (u16*)take(NTOK * C * 2);
  u16* wqb = (u16*)take(C * C * 2);
  u16* wkb = (u16*)take(C * C * 2);
  u16* wvb = (u16*)take(C * C * 2);
  u16* wpb = (u16*)take(C * C * 2);
  u16* qb  = (u16*)take(NTOK * C * 2);
  u16* kb  = (u16*)take(NTOK * C * 2);
  u16* vt  = (u16*)take(C * NTOK * 2);
  float* relh = (float*)take((size_t)12 * 64 * 64 * 64 * 4);
  float* relw = (float*)take((size_t)12 * 64 * 64 * 64 * 4);
  u16* ao  = (u16*)take(NTOK * C * 2);

  const int nx = (int)(NTOK * C), nw = (int)(C * C);
  k_cvt<<<(nx + 255) / 256, 256, 0, stream>>>(x, xb, nx);
  k_cvt<<<(nw + 255) / 256, 256, 0, stream>>>(q_w, wqb, nw);
  k_cvt<<<(nw + 255) / 256, 256, 0, stream>>>(k_w, wkb, nw);
  k_cvt<<<(nw + 255) / 256, 256, 0, stream>>>(v_w, wvb, nw);
  k_cvt<<<(nw + 255) / 256, 256, 0, stream>>>(p_w, wpb, nw);

  k_qkv<<<dim3(64, 12, 3), 128, 0, stream>>>(xb, wqb, wkb, wvb, q_b, k_b, v_b, qb, kb, vt);
  k_rel<<<dim3(64, 12), 64, 0, stream>>>(qb, rph, rpw, relh, relw);
  k_attn<<<dim3(64, 12), 128, 0, stream>>>(qb, kb, vt, relh, relw, ao);
  k_proj<<<dim3(64, 12), 128, 0, stream>>>(ao, wpb, p_b, (float*)d_out);
}
